// SLAYFeatures_27805618274331
// MI455X (gfx1250) — compile-verified, matrix operand order checked by reading, not run on hardware
//
#include <hip/hip_runtime.h>
#include <hip/hip_bf16.h>
#include <math.h>

typedef __attribute__((ext_vector_type(16))) _Float16 v16h;
typedef __attribute__((ext_vector_type(8)))  float    v8f;

namespace {
constexpr int kB = 8192;
constexpr int kH = 16;
constexpr int kD = 64;
constexpr int kP = 16;
constexpr int kM = 32;
constexpr int kR = 2;
constexpr int kWaves = 4;                     // waves per block
constexpr int kOutPerB = kR * kH * kP * kM;   // 16384

// Gauss-Laguerre(n=2) nodes/weights, scaled by 1/C, C = 2 + 1e-6
constexpr float kS[2]    = {0.29289307f, 1.70710593f};   // s_r
constexpr float kSq2S[2] = {0.76536673f, 1.84775861f};   // sqrt(2*s_r)
constexpr float kSqW[2]  = {0.65328133f, 0.27059798f};   // sqrt(w_r)
constexpr float kInvSqrtM = 0.17677669529f;              // 1/sqrt(32)
}

// 16-bit A/B fragment K-index map for v_wmma_*_16x16x32 (wave32).
// element e of the v16h: vgpr = e>>1, half = e&1.
__device__ __forceinline__ int kmap(int lane, int e) {
  int v = e >> 1;
  int hf = e & 1;
  bool lo = (lane < 16);
  int base = (v < 4) ? ((lo ? 0 : 8) + 2 * v)
                     : ((lo ? 16 : 24) + 2 * (v - 4));
  return base + hf;
}

__global__ __launch_bounds__(kWaves * 32)
void slay_features_kernel(const float* __restrict__ x,
                          const float* __restrict__ omega,
                          const float* __restrict__ anchors,
                          float* __restrict__ out) {
  __shared__ __align__(16) float s_x[kWaves][16][kD];        // 16 KB
  __shared__ float s_norm[kWaves][16];
  __shared__ __align__(16) float s_poly[kWaves][16][kP];     // 4 KB
  __shared__ __align__(16) float s_prf[kWaves][kR][16][kM];  // 16 KB

  const int lane = threadIdx.x & 31;
  const int w    = threadIdx.x >> 5;
  const int gw   = blockIdx.x * kWaves + w;   // 0 .. 8191
  const int h    = gw & (kH - 1);
  const int b0   = (gw >> 4) * 16;            // batch tile start

  // ---- cooperative load of the 16x64 x-tile into LDS (coalesced b128) ----
  const float* xbase = x + (size_t)b0 * (kH * kD) + h * kD;
#pragma unroll
  for (int i = 0; i < 8; ++i) {
    int idx = i * 32 + lane;          // float4 index, 0..255
    int row = idx >> 4;               // 16 float4 per row
    int c4  = (idx & 15) << 2;
    float4 v = *(const float4*)(xbase + (size_t)row * (kH * kD) + c4);
    *(float4*)&s_x[w][row][c4] = v;
  }
  __syncthreads();

  // ---- per-row inverse norm: 1/(sqrt(max(ss,1e-12)) + 1e-4) ----
  if (lane < 16) {
    float ss = 0.f;
#pragma unroll
    for (int d = 0; d < kD; ++d) { float t = s_x[w][lane][d]; ss += t * t; }
    float nrm = sqrtf(fmaxf(ss, 1e-12f));
    s_norm[w][lane] = 1.0f / (nrm + 1e-4f);
  }
  __syncthreads();

  // ---- build A fragments (normalized x -> f16), two K-chunks of 32 ----
  const int arow = lane & 15;
  const float inv = s_norm[w][arow];
  v16h a0, a1;
#pragma unroll
  for (int e = 0; e < 16; ++e) {
    int k = kmap(lane, e);
    a0[e] = (_Float16)(s_x[w][arow][k] * inv);
    a1[e] = (_Float16)(s_x[w][arow][k + 32] * inv);
  }

  const int nn = lane & 15;           // N column held by this lane
  const int mo = (lane >> 4) * 8;     // M row offset of C/D fragment

  // ---- poly projection: A(16x64) x anchors^T(64x16), then square/4 ----
  {
    v16h pb0, pb1;
#pragma unroll
    for (int e = 0; e < 16; ++e) {
      int k = kmap(lane, e);
      pb0[e] = (_Float16)anchors[nn * kD + k];
      pb1[e] = (_Float16)anchors[nn * kD + k + 32];
    }
    v8f acc = {};
    acc = __builtin_amdgcn_wmma_f32_16x16x32_f16(false, a0, false, pb0,
                                                 (short)0, acc, false, false);
    acc = __builtin_amdgcn_wmma_f32_16x16x32_f16(false, a1, false, pb1,
                                                 (short)0, acc, false, false);
#pragma unroll
    for (int j = 0; j < 8; ++j) {
      float pr = acc[j];
      s_poly[w][mo + j][nn] = pr * pr * 0.25f;   // /sqrt(16)
    }
  }

  // ---- PRF projections: A(16x64) x omega[r,h](64x32), exp quadrature ----
#pragma unroll
  for (int r = 0; r < kR; ++r) {
#pragma unroll
    for (int nh = 0; nh < 2; ++nh) {
      const int m = nh * 16 + nn;
      const float* ob = omega + (size_t)(r * kH + h) * kD * kM + m;
      v16h wb0, wb1;
#pragma unroll
      for (int e = 0; e < 16; ++e) {
        int k = kmap(lane, e);
        wb0[e] = (_Float16)ob[(size_t)k * kM];
        wb1[e] = (_Float16)ob[(size_t)(k + 32) * kM];
      }
      v8f acc = {};
      acc = __builtin_amdgcn_wmma_f32_16x16x32_f16(false, a0, false, wb0,
                                                   (short)0, acc, false, false);
      acc = __builtin_amdgcn_wmma_f32_16x16x32_f16(false, a1, false, wb1,
                                                   (short)0, acc, false, false);
#pragma unroll
      for (int j = 0; j < 8; ++j) {
        float arg = acc[j] * kSq2S[r] - kS[r];
        arg = fminf(fmaxf(arg, -20.f), 20.f);
        s_prf[w][r][mo + j][m] = __expf(arg) * kInvSqrtM * kSqW[r];
      }
    }
  }
  __syncthreads();

  // ---- outer product + coalesced b128 stores (bandwidth-bound stage) ----
#pragma unroll 1
  for (int row = 0; row < 16; ++row) {
#pragma unroll
    for (int r = 0; r < kR; ++r) {
      float* ob = out + (size_t)(b0 + row) * kOutPerB
                      + r * (kH * kP * kM) + h * (kP * kM);
#pragma unroll
      for (int it = 0; it < 4; ++it) {
        int idx = it * 128 + lane * 4;   // 0..508, step 4
        int p   = idx >> 5;
        int m4  = idx & 31;
        float pf = s_poly[w][row][p];
        float4 pr = *(const float4*)&s_prf[w][r][row][m4];
        float4 o;
        o.x = pf * pr.x; o.y = pf * pr.y; o.z = pf * pr.z; o.w = pf * pr.w;
        *(float4*)(ob + idx) = o;
      }
    }
  }
}

extern "C" void kernel_launch(void* const* d_in, const int* in_sizes, int n_in,
                              void* d_out, int out_size, void* d_ws, size_t ws_size,
                              hipStream_t stream) {
  const float* x       = (const float*)d_in[0];
  const float* omega   = (const float*)d_in[1];
  const float* anchors = (const float*)d_in[2];
  float* out = (float*)d_out;

  const int total_wave_tiles = (kB / 16) * kH;      // 8192
  const int blocks = total_wave_tiles / kWaves;     // 2048
  hipLaunchKernelGGL(slay_features_kernel, dim3(blocks), dim3(kWaves * 32), 0,
                     stream, x, omega, anchors, out);
}
